// SA_Layer_45097156608033
// MI455X (gfx1250) — matrix operand order repaired and checked by reading, not compiled, not run on hardware
//
#include <hip/hip_runtime.h>

// ---------------------------------------------------------------------------
// Self-attention for MI455X (gfx1250): bf16 WMMA pipeline + TDM staging.
//   K1: fused QKV projection (X tile reused for Wq/Wk/Wv), bf16 out to ws.
//   K2: flash attention (online softmax); K tile staged via Tensor Data Mover
//       (tensor_load_to_lds + s_wait_tensorcnt), V transposed by the ALUs.
// ---------------------------------------------------------------------------

typedef __attribute__((ext_vector_type(16))) __bf16 v16bf;
typedef __attribute__((ext_vector_type(8)))  float  v8f;
typedef __attribute__((ext_vector_type(4)))  unsigned int u32x4;
typedef __attribute__((ext_vector_type(8)))  int  i32x8;
typedef __attribute__((ext_vector_type(4)))  int  i32x4;

struct alignas(16) B128 { unsigned int w0, w1, w2, w3; };
union Frag32 { v16bf v; B128 q[2]; };

// A fragment (16x32 bf16, MxK): lane holds row M=lane&15; k-half base = (lane>>4)*8.
// VGPR0-3 = k[base..base+7], VGPR4-7 = k[base+16..base+23]  -> two 16B loads.
__device__ __forceinline__ v16bf load_a_frag(const __bf16* base, int pitch, int lane) {
  const __bf16* r = base + (size_t)(lane & 15) * pitch + ((lane >> 4) & 1) * 8;
  Frag32 f;
  f.q[0] = *(const B128*)(r);
  f.q[1] = *(const B128*)(r + 16);
  return f.v;
}

// B fragment (32x16 bf16, KxN) from an [N][K]-major tile: lane holds col N=lane&15;
// k-half base = (lane>>4)*8: VGPR0-3 = k[base..base+7], VGPR4-7 = k[base+16..base+23].
__device__ __forceinline__ v16bf load_b_frag(const __bf16* base, int pitch, int lane) {
  const __bf16* r = base + (size_t)(lane & 15) * pitch + ((lane >> 4) & 1) * 8;
  Frag32 f;
  f.q[0] = *(const B128*)(r);
  f.q[1] = *(const B128*)(r + 16);
  return f.v;
}

// Issue a TDM load of a 2D tile (nrows x rowbytes, contiguous rows) from global
// memory into LDS at byte offset lds_off. data_size = 8B units.
__device__ __forceinline__ void tdm_load_tile(const void* gptr, unsigned lds_off,
                                              unsigned rowbytes, unsigned nrows) {
  const unsigned long long ga = (unsigned long long)(size_t)gptr;
  const unsigned units = rowbytes >> 3;  // 8-byte units per row

  u32x4 g0;
  g0[0] = 1u;                                        // count=1, is_restore=0
  g0[1] = lds_off;                                   // lds_addr (bytes)
  g0[2] = (unsigned)(ga & 0xFFFFFFFFu);              // global_addr[31:0]
  g0[3] = (unsigned)(ga >> 32) | (2u << 30);         // global_addr[56:32] | type=2

  i32x8 g1;
  g1[0] = (int)(3u << 16);                           // data_size=3 (8B), mask=0
  g1[1] = (int)(units << 16);                        // tensor_dim0[15:0]
  g1[2] = (int)(nrows << 16);                        // dim0[31:16]=0 | tensor_dim1[15:0]
  g1[3] = (int)(units << 16);                        // dim1[31:16]=0 | tile_dim0
  g1[4] = (int)nrows;                                // tile_dim1 | tile_dim2=0
  g1[5] = (int)units;                                // tensor_dim0_stride[31:0]
  g1[6] = 0;                                         // stride[47:32] | dim1_stride lo
  g1[7] = 0;

  i32x4 gz;
  gz[0] = 0; gz[1] = 0; gz[2] = 0; gz[3] = 0;

#if __has_include(<hip/amd_detail/amd_gfx1250_TDM.h>)
  i32x8 gz8;
  for (int i = 0; i < 8; ++i) gz8[i] = 0;
  __builtin_amdgcn_tensor_load_to_lds(g0, g1, gz, gz, gz8, 0);
#else
  __builtin_amdgcn_tensor_load_to_lds(g0, g1, gz, gz, 0);
#endif
}

// ---------------------------------------------------------------------------
// Kernel 1: q/k/v = x @ W{q,k,v} + b{q,k,v}   (M=16384, N=512, K=512)
// Block = 256 threads (8 waves), 64x64 output tile, all three outputs at once.
// ---------------------------------------------------------------------------
__global__ __launch_bounds__(256)
void qkv_proj_kernel(const float* __restrict__ x,
                     const float* __restrict__ Wq, const float* __restrict__ bq,
                     const float* __restrict__ Wk, const float* __restrict__ bk,
                     const float* __restrict__ Wv, const float* __restrict__ bv,
                     __bf16* __restrict__ qo, __bf16* __restrict__ ko,
                     __bf16* __restrict__ vo)
{
  __shared__ alignas(16) __bf16 Xs[64 * 32];      // [m][k]
  __shared__ alignas(16) __bf16 Ws[3][64 * 32];   // [n][k] (transposed weight tile)

  const int tid  = threadIdx.x;
  const int lane = tid & 31;
  const int wid  = tid >> 5;
  const int wm   = wid >> 2;        // 0..1 : 32-row slab
  const int wn   = wid & 3;         // 0..3 : 16-col slab
  const int bm   = blockIdx.x >> 3; // 256 row-blocks
  const int bn   = blockIdx.x & 7;  // 8 col-blocks
  const int row0 = bm * 64, col0 = bn * 64;

  const float* Wp[3] = {Wq, Wk, Wv};

  v8f acc[3][2];
  for (int w = 0; w < 3; ++w)
    for (int mt = 0; mt < 2; ++mt)
      for (int e = 0; e < 8; ++e) acc[w][mt][e] = 0.0f;

  for (int k0 = 0; k0 < 512; k0 += 32) {
    // --- stage X tile (64x32), fp32 -> bf16 ---
    {
      const int r = tid >> 2, c = (tid & 3) * 8;
      const float* src = x + (size_t)(row0 + r) * 512 + k0 + c;
      float4 f0 = ((const float4*)src)[0];
      float4 f1 = ((const float4*)src)[1];
      __bf16* dst = &Xs[r * 32 + c];
      dst[0] = (__bf16)f0.x; dst[1] = (__bf16)f0.y;
      dst[2] = (__bf16)f0.z; dst[3] = (__bf16)f0.w;
      dst[4] = (__bf16)f1.x; dst[5] = (__bf16)f1.y;
      dst[6] = (__bf16)f1.z; dst[7] = (__bf16)f1.w;
    }
    // --- stage W tiles transposed: Ws[w][n][kk] = W[k0+kk][col0+n] ---
    {
      const int n = tid & 63, kk = (tid >> 6) * 8;
      for (int w = 0; w < 3; ++w) {
        const float* W = Wp[w];
        __bf16* dst = &Ws[w][n * 32 + kk];
        for (int j = 0; j < 8; ++j)
          dst[j] = (__bf16)W[(size_t)(k0 + kk + j) * 512 + col0 + n];
      }
    }
    __syncthreads();

    v16bf a0 = load_a_frag(&Xs[(wm * 32 +  0) * 32], 32, lane);
    v16bf a1 = load_a_frag(&Xs[(wm * 32 + 16) * 32], 32, lane);
    for (int w = 0; w < 3; ++w) {
      v16bf bf = load_b_frag(&Ws[w][(wn * 16) * 32], 32, lane);
      acc[w][0] = __builtin_amdgcn_wmma_f32_16x16x32_bf16(
          false, a0, false, bf, (short)0, acc[w][0], false, false);
      acc[w][1] = __builtin_amdgcn_wmma_f32_16x16x32_bf16(
          false, a1, false, bf, (short)0, acc[w][1], false, false);
    }
    __syncthreads();
  }

  // --- epilogue: +bias, store bf16 ---
  const int hl = (lane >> 4) & 1;
  const int gn = col0 + wn * 16 + (lane & 15);
  const float* bp[3]  = {bq, bk, bv};
  __bf16*      op[3]  = {qo, ko, vo};
  for (int w = 0; w < 3; ++w) {
    const float bias = bp[w][gn];
    for (int mt = 0; mt < 2; ++mt) {
      const int gm0 = row0 + wm * 32 + mt * 16 + 8 * hl;
      for (int i = 0; i < 8; ++i)
        op[w][(size_t)(gm0 + i) * 512 + gn] = (__bf16)(acc[w][mt][i] + bias);
    }
  }
}

// ---------------------------------------------------------------------------
// Kernel 2: flash attention. Grid = 8 batches * 64 s-blocks of 32 rows.
// 8 waves: rt = wid>>2 picks 16-row tile, widx = wid&3 picks S col-tile
// (16 of 64 keys) and O d-chunk (128 of 512). Online softmax; m/l replicated
// per wave (identical LDS reductions keep them consistent).
// K tile is DMA'd to LDS by wave 0 via the Tensor Data Mover while the other
// waves transpose V into LDS.
// ---------------------------------------------------------------------------
__global__ __launch_bounds__(256)
void flash_attn_kernel(const __bf16* __restrict__ qm,
                       const __bf16* __restrict__ km,
                       const __bf16* __restrict__ vm,
                       float* __restrict__ out)
{
  extern __shared__ char smem_raw[];
  __bf16* Ks   = (__bf16*)smem_raw;           // [64][512]   64 KB  (LDS byte off 0)
  __bf16* Vt   = Ks + 64 * 512;               // [512][64]   64 KB  (transposed)
  __bf16* Ps   = Vt + 512 * 64;               // [2][16][64]  4 KB
  float*  rmax = (float*)(Ps + 2 * 16 * 64);  // [2][4][16]
  float*  rsum = rmax + 2 * 4 * 16;           // [2][4][16]

  const int tid  = threadIdx.x;
  const int lane = tid & 31;
  const int wid  = tid >> 5;
  const int rt   = wid >> 2;       // 0..1
  const int widx = wid & 3;        // 0..3
  const int hl   = (lane >> 4) & 1;
  const int ln   = lane & 15;

  const int b  = blockIdx.x >> 6;
  const int sb = blockIdx.x & 63;

  const size_t qoff  = ((size_t)b * 2048 + (size_t)sb * 32) * 512;
  const size_t kvoff = (size_t)b * 2048 * 512;
  const __bf16* qrow = qm + qoff + (size_t)(rt * 16) * 512;  // this wave's 16 q rows

  v8f acc[8];
  for (int nt = 0; nt < 8; ++nt)
    for (int e = 0; e < 8; ++e) acc[nt][e] = 0.0f;
  float mrow[8], lrow[8];
  for (int i = 0; i < 8; ++i) { mrow[i] = -1e30f; lrow[i] = 0.0f; }

  for (int t0 = 0; t0 < 2048; t0 += 64) {
    __syncthreads();  // LDS reuse barrier (prev iter's GEMMs read Ks/Vt/Ps)

    // --- stage K tile [64][512] via Tensor Data Mover (wave 0 issues) ---
    if (wid == 0) {
      tdm_load_tile(km + kvoff + (size_t)t0 * 512, /*lds_off=*/0u,
                    /*rowbytes=*/1024u, /*nrows=*/64u);
    }
    // --- stage V tile transposed: Vt[d][t] = v[t0+t][d] (all waves) ---
    {
      const __bf16* vsrc = vm + kvoff + (size_t)t0 * 512;
      const bool more = (t0 + 64) < 2048;
      for (int i = tid; i < 64 * 64; i += 256) {
        const int t = i & 63, d8 = i >> 6;
        const __bf16* sp = vsrc + (size_t)t * 512 + d8 * 8;
        B128 pk = *(const B128*)sp;
        if (more) __builtin_prefetch(sp + 64 * 512, 0, 1);
        const __bf16* e = (const __bf16*)&pk;
        __bf16* dcol = &Vt[(size_t)(d8 * 8) * 64 + t];
        for (int j = 0; j < 8; ++j) dcol[j * 64] = e[j];
      }
    }
    if (wid == 0) __builtin_amdgcn_s_wait_tensorcnt(0);
    __syncthreads();

    // --- S[16x16] = Q(rt) x K(widx)^T over d=512 ---
    v8f s;
    for (int e = 0; e < 8; ++e) s[e] = 0.0f;
    for (int k0 = 0; k0 < 512; k0 += 32) {
      v16bf a  = load_a_frag(qrow + k0, 512, lane);                       // global, L2-hot
      v16bf bf = load_b_frag(Ks + (size_t)(widx * 16) * 512 + k0, 512, lane);
      s = __builtin_amdgcn_wmma_f32_16x16x32_bf16(
          false, a, false, bf, (short)0, s, false, false);
    }

    // --- per-row max of this tile (16-lane halves are independent rows) ---
    float pmax[8];
    for (int i = 0; i < 8; ++i) {
      float m = s[i];
      m = fmaxf(m, __shfl_xor(m, 1, 32));
      m = fmaxf(m, __shfl_xor(m, 2, 32));
      m = fmaxf(m, __shfl_xor(m, 4, 32));
      m = fmaxf(m, __shfl_xor(m, 8, 32));
      pmax[i] = m;
    }
    if (ln == 0)
      for (int i = 0; i < 8; ++i)
        rmax[(rt * 4 + widx) * 16 + i + 8 * hl] = pmax[i];
    __syncthreads();

    // --- combine across 4 col-tiles; online rescale factors ---
    float mnew[8], scl[8];
    for (int i = 0; i < 8; ++i) {
      const int r = i + 8 * hl;
      float m = mrow[i];
      for (int j = 0; j < 4; ++j) m = fmaxf(m, rmax[(rt * 4 + j) * 16 + r]);
      mnew[i] = m;
      scl[i]  = __expf(mrow[i] - m);
    }

    // --- P = exp(S - mnew): stage to LDS (bf16, A-layout source), row sums ---
    float psum[8];
    for (int i = 0; i < 8; ++i) {
      float p = __expf(s[i] - mnew[i]);
      Ps[(size_t)(rt * 16 + i + 8 * hl) * 64 + widx * 16 + ln] = (__bf16)p;
      float t = p;
      t += __shfl_xor(t, 1, 32);
      t += __shfl_xor(t, 2, 32);
      t += __shfl_xor(t, 4, 32);
      t += __shfl_xor(t, 8, 32);
      psum[i] = t;
    }
    if (ln == 0)
      for (int i = 0; i < 8; ++i)
        rsum[(rt * 4 + widx) * 16 + i + 8 * hl] = psum[i];
    __syncthreads();

    // --- update l, rescale O accumulators ---
    for (int i = 0; i < 8; ++i) {
      const int r = i + 8 * hl;
      float ssum = 0.0f;
      for (int j = 0; j < 4; ++j) ssum += rsum[(rt * 4 + j) * 16 + r];
      lrow[i] = lrow[i] * scl[i] + ssum;
      mrow[i] = mnew[i];
      for (int nt = 0; nt < 8; ++nt) acc[nt][i] *= scl[i];
    }

    // --- O(d-chunk widx*128) += P[16x64] x V[64x128] ---
    for (int ks = 0; ks < 2; ++ks) {
      v16bf a = load_a_frag(Ps + (size_t)(rt * 16) * 64 + ks * 32, 64, lane);
      for (int nt = 0; nt < 8; ++nt) {
        v16bf bf = load_b_frag(
            Vt + (size_t)(widx * 128 + nt * 16) * 64 + ks * 32, 64, lane);
        acc[nt] = __builtin_amdgcn_wmma_f32_16x16x32_bf16(
            false, a, false, bf, (short)0, acc[nt], false, false);
      }
    }
  }

  // --- epilogue: O / l, fp32 out ---
  const size_t orow0 = (size_t)b * 2048 + (size_t)sb * 32 + rt * 16 + 8 * hl;
  for (int nt = 0; nt < 8; ++nt) {
    const int gd = widx * 128 + nt * 16 + ln;
    for (int i = 0; i < 8; ++i)
      out[(orow0 + i) * 512 + gd] = acc[nt][i] / lrow[i];
  }
}

// ---------------------------------------------------------------------------
extern "C" void kernel_launch(void* const* d_in, const int* in_sizes, int n_in,
                              void* d_out, int out_size, void* d_ws, size_t ws_size,
                              hipStream_t stream) {
  (void)in_sizes; (void)n_in; (void)out_size; (void)ws_size;
  const float* x  = (const float*)d_in[0];
  const float* Wq = (const float*)d_in[1];
  const float* bq = (const float*)d_in[2];
  const float* Wk = (const float*)d_in[3];
  const float* bk = (const float*)d_in[4];
  const float* Wv = (const float*)d_in[5];
  const float* bv = (const float*)d_in[6];
  float* out = (float*)d_out;

  const size_t elems = (size_t)8 * 2048 * 512;   // 8,388,608 per tensor
  __bf16* q = (__bf16*)d_ws;                      // 16 MB each in ws (48 MB total)
  __bf16* k = q + elems;
  __bf16* v = k + elems;

  // K1: 16384/64 row-blocks * 512/64 col-blocks = 2048 blocks
  qkv_proj_kernel<<<dim3(2048), dim3(256), 0, stream>>>(
      x, Wq, bq, Wk, bk, Wv, bv, q, k, v);

  // K2: 8 batches * 64 s-blocks = 512 blocks; 133 KB dynamic LDS
  const size_t smem = (size_t)(64 * 512 + 512 * 64 + 2 * 16 * 64) * sizeof(__bf16)
                    + (size_t)(2 * 2 * 4 * 16) * sizeof(float);
  flash_attn_kernel<<<dim3(512), dim3(256), smem, stream>>>(q, k, v, out);
}
